// PhysicalConsistencyLoss_39651138077317
// MI455X (gfx1250) — matrix operand order repaired and checked by reading, not compiled, not run on hardware
//
#include <hip/hip_runtime.h>

#define B_DIM 16384
#define Z_DIM 2048
#define NUM_ROWBLK (B_DIM / 128)   // 128
#define NUM_ZBLK   (Z_DIM / 64)    // 32
#define NUM_WG     (NUM_ROWBLK * NUM_ZBLK)
#define NK         (Z_DIM / 32)    // 64 K-steps

static constexpr float BETA       = 0.3f;
static constexpr float LAMBDA_PHY = 0.15f;

typedef __attribute__((ext_vector_type(16))) _Float16 v16h;
typedef __attribute__((ext_vector_type(8)))  _Float16 v8h;
typedef __attribute__((ext_vector_type(8)))  float    v8f;
typedef __attribute__((ext_vector_type(4)))  float    v4f;
typedef __attribute__((ext_vector_type(4)))  int      v4i;

// gfx1250 async global->LDS path (ASYNCcnt), with compile-safe fallback.
#if __has_builtin(__builtin_amdgcn_global_load_async_to_lds_b128) && \
    __has_builtin(__builtin_amdgcn_s_wait_asynccnt)
#define USE_ASYNC_LDS 1
#else
#define USE_ASYNC_LDS 0
#endif

// Load a 16-element f16 fragment from LDS as two 16B-aligned halves.
__device__ __forceinline__ v16h ld_frag(const _Float16* p) {
  v8h a = *(const v8h*)p;
  v8h b = *(const v8h*)(p + 8);
  v16h r;
#pragma unroll
  for (int i = 0; i < 8; ++i) { r[i] = a[i]; r[i + 8] = b[i]; }
  return r;
}

// ---------------------------------------------------------------------------
// Kernel 1: adjacency -> f16 mask, per-zone 1/deg and has-neighbor flag.
// ---------------------------------------------------------------------------
__global__ __launch_bounds__(256) void prep_mask_kernel(
    const int* __restrict__ adj, _Float16* __restrict__ maskh,
    float* __restrict__ invdeg, float* __restrict__ nbf) {
  __shared__ float sRed[8];
  const int z = blockIdx.x;
  const int tid = threadIdx.x;
  float cnt = 0.f;
  for (int c = tid; c < Z_DIM; c += 256) {
    const int a = adj[(size_t)z * Z_DIM + c];
    const float m = (a > 0) ? 1.f : 0.f;
    maskh[(size_t)z * Z_DIM + c] = (_Float16)m;
    cnt += m;
  }
#pragma unroll
  for (int o = 16; o > 0; o >>= 1) cnt += __shfl_down(cnt, o, 32);
  if ((tid & 31) == 0) sRed[tid >> 5] = cnt;
  __syncthreads();
  if (tid == 0) {
    float d = 0.f;
#pragma unroll
    for (int i = 0; i < 8; ++i) d += sRed[i];
    invdeg[z] = (d > 0.f) ? 1.f / d : 1.f;
    nbf[z]    = (d > 0.f) ? 1.f : 0.f;
  }
}

// ---------------------------------------------------------------------------
// Kernel 2: fused split-f16 WMMA GEMM (avg_neighbor) + softplus physics loss
// + smooth-L1 pred loss.  Macro tile 128(B) x 64(Z), 8 waves, double-buffered
// LDS so WMMA of step k overlaps staging of step k+1.
// ---------------------------------------------------------------------------
__global__ __launch_bounds__(256) void fused_gemm_loss_kernel(
    const float* __restrict__ preds, const float* __restrict__ targets,
    const float* __restrict__ temps, const _Float16* __restrict__ maskh,
    const float* __restrict__ invdeg, const float* __restrict__ nbf,
    float* __restrict__ partial_pred, float* __restrict__ partial_phys) {
  // Padded stride 40 f16 (= 80B) => bank-conflict-free b128 reads.
  __shared__ _Float16 sAhi[2][128 * 40];
  __shared__ _Float16 sAlo[2][128 * 40];
  __shared__ _Float16 sB[2][64 * 40];
  __shared__ float sRed[16];

  const int tid  = threadIdx.x;
  const int lane = tid & 31;
  const int w    = tid >> 5;       // wave id 0..7
  const int l15  = lane & 15;
  const int sel  = lane >> 4;      // fragment half select

  const int rowBase = blockIdx.y * 128;  // B block (x-fastest grid keeps T hot in L2)
  const int zBase   = blockIdx.x * 64;   // Z block

  // staging roles
  const int aRow  = tid >> 1;      // 0..127
  const int aHalf = tid & 1;       // K half (16 elems)
  const int bRow  = tid >> 2;      // 0..63
  const int bQ    = tid & 3;       // 8-elem quarter of 32-K row

  const float*    gA0 = temps + (size_t)(rowBase + aRow) * Z_DIM + aHalf * 16;
  const _Float16* gB0 = maskh + (size_t)(zBase + bRow) * Z_DIM + bQ * 8;

  v8f acc[4];
#pragma unroll
  for (int t = 0; t < 4; ++t) acc[t] = (v8f)0.0f;

  // ---- tile staging: A (f32 -> hi/lo f16, K kept contiguous) + B copy
  auto stage = [&](int k, int buf) {
    const float* gA = gA0 + k;
#pragma unroll
    for (int g = 0; g < 2; ++g) {
      v4f f0 = *(const v4f*)(gA + g * 8);
      v4f f1 = *(const v4f*)(gA + g * 8 + 4);
      v8h h, l;
#pragma unroll
      for (int i = 0; i < 4; ++i) {
        _Float16 hh = (_Float16)f0[i];
        h[i] = hh; l[i] = (_Float16)(f0[i] - (float)hh);
        _Float16 hh2 = (_Float16)f1[i];
        h[4 + i] = hh2; l[4 + i] = (_Float16)(f1[i] - (float)hh2);
      }
      // K chunk stored contiguously: LDS element index == K within the step
      const int off = aRow * 40 + aHalf * 16 + g * 8;
      *(v8h*)(&sAhi[buf][off]) = h;
      *(v8h*)(&sAlo[buf][off]) = l;
    }
    const _Float16* gB = gB0 + k;
    _Float16* dB = &sB[buf][bRow * 40 + bQ * 8];
#if USE_ASYNC_LDS
    __builtin_amdgcn_global_load_async_to_lds_b128(
        (__attribute__((address_space(1))) v4i*)gB,
        (__attribute__((address_space(3))) v4i*)dB, 0, 0);
#else
    *(v8h*)dB = *(const v8h*)gB;
#endif
    if (k + 32 < Z_DIM) {                 // -> global_prefetch_b8
      __builtin_prefetch(gA + 32, 0, 1);
      __builtin_prefetch(gB + 32, 0, 1);
    }
  };

  stage(0, 0);
  for (int kk = 0; kk < NK; ++kk) {
#if USE_ASYNC_LDS
    __builtin_amdgcn_s_wait_asynccnt(0);  // async LDS writes visible pre-barrier
#endif
    __syncthreads();
    if (kk + 1 < NK) stage((kk + 1) * 32, (kk + 1) & 1);

    const int buf = kk & 1;
    const v16h aHi = ld_frag(&sAhi[buf][(w * 16 + l15) * 40 + sel * 16]);
    const v16h aLo = ld_frag(&sAlo[buf][(w * 16 + l15) * 40 + sel * 16]);
#pragma unroll
    for (int t = 0; t < 4; ++t) {
      const v16h b = ld_frag(&sB[buf][(t * 16 + l15) * 40 + sel * 16]);
      acc[t] = __builtin_amdgcn_wmma_f32_16x16x32_f16(
          false, aHi, false, b, (short)0, acc[t], false, false);
      acc[t] = __builtin_amdgcn_wmma_f32_16x16x32_f16(
          false, aLo, false, b, (short)0, acc[t], false, false);
    }
  }

  // ---- fused epilogue: C/D layout -> lane holds col N=l15, rows sel*8+j
  float predAcc = 0.f, physAcc = 0.f;
#pragma unroll
  for (int t = 0; t < 4; ++t) {
    const int z   = zBase + t * 16 + l15;
    const float inv = invdeg[z];
    const float nbz = nbf[z];
    const int r0 = rowBase + w * 16 + sel * 8;
#pragma unroll
    for (int j = 0; j < 8; ++j) {
      const size_t idx = (size_t)(r0 + j) * Z_DIM + z;
      const float p  = preds[idx];
      const float tg = targets[idx];
      const float ct = temps[idx];
      const float avg = acc[t][j] * inv;
      const float x = (ct - avg) * (p - ct);
      const float sp = fmaxf(x, 0.f) + log1pf(expf(-fabsf(x)));  // stable softplus
      physAcc += nbz * sp;
      const float d = p - tg;
      const float ad = fabsf(d);
      predAcc += (ad < BETA) ? (0.5f / BETA) * d * d : (ad - 0.5f * BETA);
    }
  }
#pragma unroll
  for (int o = 16; o > 0; o >>= 1) {
    predAcc += __shfl_down(predAcc, o, 32);
    physAcc += __shfl_down(physAcc, o, 32);
  }
  if (lane == 0) { sRed[w] = predAcc; sRed[8 + w] = physAcc; }
  __syncthreads();
  if (tid == 0) {
    float ps = 0.f, hs = 0.f;
#pragma unroll
    for (int i = 0; i < 8; ++i) { ps += sRed[i]; hs += sRed[8 + i]; }
    const int wg = blockIdx.y * gridDim.x + blockIdx.x;
    partial_pred[wg] = ps;
    partial_phys[wg] = hs;
  }
}

// ---------------------------------------------------------------------------
// Kernel 3: deterministic reduction of per-WG partials, emit both losses.
// ---------------------------------------------------------------------------
__global__ __launch_bounds__(256) void finalize_kernel(
    const float* __restrict__ pp, const float* __restrict__ ph,
    float* __restrict__ out, int n) {
  __shared__ float sRed[16];
  const int tid = threadIdx.x;
  float a = 0.f, b = 0.f;
  for (int i = tid; i < n; i += 256) { a += pp[i]; b += ph[i]; }
#pragma unroll
  for (int o = 16; o > 0; o >>= 1) {
    a += __shfl_down(a, o, 32);
    b += __shfl_down(b, o, 32);
  }
  if ((tid & 31) == 0) { sRed[tid >> 5] = a; sRed[8 + (tid >> 5)] = b; }
  __syncthreads();
  if (tid == 0) {
    float ps = 0.f, hs = 0.f;
#pragma unroll
    for (int i = 0; i < 8; ++i) { ps += sRed[i]; hs += sRed[8 + i]; }
    const float invN = 1.f / ((float)B_DIM * (float)Z_DIM);
    const float pred_loss = ps * invN;
    const float phys_loss = hs * invN;   // sum/(B*Z) == masked per-zone mean / Z
    out[0] = pred_loss + LAMBDA_PHY * phys_loss;
    out[1] = phys_loss;
  }
}

// ---------------------------------------------------------------------------
extern "C" void kernel_launch(void* const* d_in, const int* in_sizes, int n_in,
                              void* d_out, int out_size, void* d_ws, size_t ws_size,
                              hipStream_t stream) {
  (void)in_sizes; (void)n_in; (void)out_size; (void)ws_size;
  const float* preds   = (const float*)d_in[0];
  const float* targets = (const float*)d_in[1];
  const float* temps   = (const float*)d_in[2];
  const int*   adj     = (const int*)d_in[3];
  float* out = (float*)d_out;

  // Workspace layout (~8.4 MB, fully rewritten every launch):
  char* ws = (char*)d_ws;
  float*    invdeg = (float*)(ws + 0);         // 2048 f32
  float*    nbf    = (float*)(ws + 8192);      // 2048 f32
  float*    pp     = (float*)(ws + 16384);     // 4096 f32 partial pred sums
  float*    ph     = (float*)(ws + 32768);     // 4096 f32 partial phys sums
  _Float16* maskh  = (_Float16*)(ws + 49152);  // 2048*2048 f16 mask

  prep_mask_kernel<<<Z_DIM, 256, 0, stream>>>(adj, maskh, invdeg, nbf);

  dim3 grid(NUM_ZBLK, NUM_ROWBLK);  // x = z-blocks fastest => T rows stay hot in L2
  fused_gemm_loss_kernel<<<grid, 256, 0, stream>>>(
      preds, targets, temps, maskh, invdeg, nbf, pp, ph);

  finalize_kernel<<<1, 256, 0, stream>>>(pp, ph, out, NUM_WG);
}
